// TransitionUp_65154653880708
// MI455X (gfx1250) — matrix operand order repaired and checked by reading, not compiled, not run on hardware
//
#include <hip/hip_runtime.h>

#define NPTS 262144
#define CCH  256
#define NSEG 16
#define BN_EPS 1e-5f

typedef __bf16 bf16;
typedef __attribute__((ext_vector_type(16))) __bf16 v16bf;
typedef __attribute__((ext_vector_type(8)))  float  v8f;

union FragAB {
  v16bf v;
  uint4 q[2];
  bf16  h[16];
};

// Workspace layout (bytes)
#define WS_SEGSUM   0                         // float[16*256]    16384
#define WS_STATSUM  16384                     // float[512]        2048
#define WS_GG       18432                     // float[16*256]    16384
#define WS_SCALE    34816                     // float[256]        1024
#define WS_SHIFT    35840                     // float[256]        1024
#define WS_W1ABF    36864                     // bf16[256*256]   131072
#define WS_XBF      167936                    // bf16[N*256]  134217728
#define WS_NEED_CACHE (167936ull + (unsigned long long)NPTS * CCH * 2ull)

// LDS layout for gemm_pass (dynamic)
#define SMX_STRIDE  264                       // 256 + 8 bf16 pad -> conflict-free b128 reads
#define SMX_BYTES   (64 * SMX_STRIDE * 2)     // 33792
#define SMEM_BYTES  (SMX_BYTES + NSEG*CCH*4 + 512*4 + 64*4)  // 52480

// ---------------------------------------------------------------------------
// Kernel 1: per-segment column sums of x (+ optional bf16 cache of x).
// ---------------------------------------------------------------------------
template<bool CACHE>
__global__ __launch_bounds__(256)
void segsum_kernel(const float* __restrict__ x, const int* __restrict__ o,
                   float* __restrict__ segsum, bf16* __restrict__ xbf)
{
  __shared__ int ssg[128];
  const int c  = threadIdx.x;
  const int r0 = blockIdx.x * 128;
  if (c < 128) {
    int row = r0 + c, s = 0;
    while (row >= o[s]) ++s;       // searchsorted(o, row, 'right')
    ssg[c] = s;
  }
  __syncthreads();
  float acc = 0.f;
  int cur = ssg[0];
  for (int i = 0; i < 128; ++i) {
    const int row = r0 + i;
    float v = x[(size_t)row * CCH + c];
    if (CACHE) xbf[(size_t)row * CCH + c] = (bf16)v;
    const int s = ssg[i];
    if (s != cur) { atomicAdd(&segsum[cur * CCH + c], acc); acc = 0.f; cur = s; }
    acc += v;
  }
  atomicAdd(&segsum[cur * CCH + c], acc);
}

// ---------------------------------------------------------------------------
// Kernel 2 (tiny, 1 block): means -> g = relu(means@w2.T+b2)
//                           -> gg[b] = g[b]@w1b.T + b1  (per-segment bias table)
//                           -> optional bf16 conversion of w1a.
// ---------------------------------------------------------------------------
template<bool CACHE>
__global__ __launch_bounds__(256)
void prep_kernel(const float* __restrict__ segsum, const float* __restrict__ w2,
                 const float* __restrict__ b2, const float* __restrict__ w1,
                 const float* __restrict__ b1, const int* __restrict__ o,
                 float* __restrict__ gg, bf16* __restrict__ w1abf)
{
  __shared__ float sm[NSEG * CCH];   // means
  __shared__ float sg[NSEG * CCH];   // g
  const int c = threadIdx.x;
  int prev = 0;
  #pragma unroll
  for (int b = 0; b < NSEG; ++b) {
    const int ob = o[b];
    const float cnt = (float)(ob - prev);
    prev = ob;
    sm[b * CCH + c] = segsum[b * CCH + c] / cnt;
  }
  __syncthreads();
  float a[NSEG];
  #pragma unroll
  for (int b = 0; b < NSEG; ++b) a[b] = b2[c];
  for (int k = 0; k < CCH; ++k) {
    const float wv = w2[(size_t)c * CCH + k];
    #pragma unroll
    for (int b = 0; b < NSEG; ++b) a[b] = fmaf(sm[b * CCH + k], wv, a[b]);
  }
  #pragma unroll
  for (int b = 0; b < NSEG; ++b) sg[b * CCH + c] = fmaxf(a[b], 0.f);
  __syncthreads();
  #pragma unroll
  for (int b = 0; b < NSEG; ++b) a[b] = b1[c];
  for (int k = 0; k < CCH; ++k) {
    const float wv = w1[(size_t)c * (2 * CCH) + CCH + k];  // w1b
    #pragma unroll
    for (int b = 0; b < NSEG; ++b) a[b] = fmaf(sg[b * CCH + k], wv, a[b]);
  }
  #pragma unroll
  for (int b = 0; b < NSEG; ++b) gg[b * CCH + c] = a[b];
  if (CACHE) {
    for (int i = threadIdx.x; i < CCH * CCH; i += 256)
      w1abf[i] = (bf16)w1[(size_t)(i >> 8) * (2 * CCH) + (i & 255)];
  }
}

// ---------------------------------------------------------------------------
// GEMM passes: h = x @ w1a.T + gg[seg]  via v_wmma_f32_16x16x32_bf16.
// Block = 8 waves x 64 rows x 256 channels; wave owns 2 N-tiles.
// Loop nest: kk OUTER, mt INNER -> each B fragment is loaded exactly once
// per wave (short live range, no rematerialization), while the persistent
// state is the 4x2 accumulator tile (64 VGPRs) -- register-friendly.
// Pass A (WRITE=0): per-channel sum(h), sum(h^2) -> LDS ds_add + global atomics.
// Pass B (WRITE=1): out = relu(h*scale + shift).
// ---------------------------------------------------------------------------
template<bool CACHE, bool WRITE>
__global__ __launch_bounds__(256)
void gemm_pass(const float* __restrict__ x, const bf16* __restrict__ xbf,
               const float* __restrict__ w1, const bf16* __restrict__ w1abf,
               const float* __restrict__ gg, const int* __restrict__ o,
               const float* __restrict__ scale, const float* __restrict__ shiftv,
               float* __restrict__ statsum, float* __restrict__ out)
{
  extern __shared__ char smraw[];
  bf16*  smx  = (bf16*)smraw;                     // [64][264] bf16
  float* sgg  = (float*)(smraw + SMX_BYTES);      // [16][256]
  float* sred = sgg + NSEG * CCH;                 // [512]
  int*   ssg  = (int*)(sred + 512);               // [64] segment id per row

  const int tid  = threadIdx.x;
  const int lane = tid & 31;
  const int w    = tid >> 5;        // wave 0..7
  const int hi   = lane >> 4;       // half-wave
  const int l15  = lane & 15;
  const int rowBase = blockIdx.x * 64;

  for (int i = tid; i < NSEG * CCH; i += 256) sgg[i] = gg[i];
  if (!WRITE) { sred[tid] = 0.f; sred[tid + 256] = 0.f; }
  if (tid < 64) {
    int row = rowBase + tid, s = 0;
    while (row >= o[s]) ++s;
    ssg[tid] = s;
  }
  if (CACHE) {
    for (int i = tid; i < (64 * CCH) / 8; i += 256) {   // 8 bf16 per chunk
      const int r = i >> 5, c8 = (i & 31) * 8;
      *(uint4*)(smx + r * SMX_STRIDE + c8) =
          *(const uint4*)(xbf + (size_t)(rowBase + r) * CCH + c8);
    }
  } else {
    for (int i = tid; i < (64 * CCH) / 8; i += 256) {
      const int r = i >> 5, c8 = (i & 31) * 8;
      const float* src = x + (size_t)(rowBase + r) * CCH + c8;
      bf16* dst = smx + r * SMX_STRIDE + c8;
      #pragma unroll
      for (int j = 0; j < 8; ++j) dst[j] = (bf16)src[j];
    }
  }
  __syncthreads();

  const int c0 = (w * 2) * 16 + l15;   // channel of N-tile 0 for this lane
  const int c1 = c0 + 16;              // channel of N-tile 1

  // Persistent accumulators: 4 M-tiles x 2 N-tiles x 8 VGPRs
  v8f acc[4][2];
  #pragma unroll
  for (int mt = 0; mt < 4; ++mt) {
    acc[mt][0] = (v8f){};
    acc[mt][1] = (v8f){};
  }

  #pragma unroll
  for (int kk = 0; kk < 8; ++kk) {
    // B fragments for this K-chunk: B[k][c] = w1a[c][k]; lane l15 -> column c.
    // Lanes 0-15 hold K 0..15 of the 32-chunk, lanes 16-31 hold K 16..31.
    FragAB B0, B1;
    {
      const int kb = kk * 32 + hi * 16;
      if (CACHE) {
        B0.q[0] = *(const uint4*)(w1abf + (size_t)c0 * CCH + kb);
        B0.q[1] = *(const uint4*)(w1abf + (size_t)c0 * CCH + kb + 8);
        B1.q[0] = *(const uint4*)(w1abf + (size_t)c1 * CCH + kb);
        B1.q[1] = *(const uint4*)(w1abf + (size_t)c1 * CCH + kb + 8);
      } else {
        const float* s0 = w1 + (size_t)c0 * (2 * CCH) + kb;
        const float* s1 = w1 + (size_t)c1 * (2 * CCH) + kb;
        #pragma unroll
        for (int j = 0; j < 16; ++j) { B0.h[j] = (bf16)s0[j]; B1.h[j] = (bf16)s1[j]; }
      }
    }
    #pragma unroll
    for (int mt = 0; mt < 4; ++mt) {
      // A fragment: lane l15 -> row; lanes 0-15: K 0-15, lanes 16-31: K 16-31
      FragAB Af;
      const int ks = kk * 32 + hi * 16;
      const bf16* ap = smx + (mt * 16 + l15) * SMX_STRIDE + ks;
      Af.q[0] = *(const uint4*)(ap);
      Af.q[1] = *(const uint4*)(ap + 8);
      acc[mt][0] = __builtin_amdgcn_wmma_f32_16x16x32_bf16(false, Af.v, false, B0.v,
                                                           (short)0, acc[mt][0], false, false);
      acc[mt][1] = __builtin_amdgcn_wmma_f32_16x16x32_bf16(false, Af.v, false, B1.v,
                                                           (short)0, acc[mt][1], false, false);
    }
  }

  float sc0 = 0.f, sc1 = 0.f, sh0 = 0.f, sh1 = 0.f;
  if (WRITE) {
    sc0 = scale[c0]; sh0 = shiftv[c0];
    sc1 = scale[c1]; sh1 = shiftv[c1];
  }
  float s10 = 0.f, s20 = 0.f, s11 = 0.f, s21 = 0.f;

  // Epilogue: D layout -> lane holds column c, VGPR r -> row (r + 8*hi)
  #pragma unroll
  for (int mt = 0; mt < 4; ++mt) {
    #pragma unroll
    for (int r = 0; r < 8; ++r) {
      const int lr  = mt * 16 + hi * 8 + r;
      const int row = rowBase + lr;
      const int sgm = ssg[lr];
      const float h0 = acc[mt][0][r] + sgg[sgm * CCH + c0];
      const float h1 = acc[mt][1][r] + sgg[sgm * CCH + c1];
      if (WRITE) {
        out[(size_t)row * CCH + c0] = fmaxf(fmaf(h0, sc0, sh0), 0.f);
        out[(size_t)row * CCH + c1] = fmaxf(fmaf(h1, sc1, sh1), 0.f);
      } else {
        s10 += h0; s20 += h0 * h0;
        s11 += h1; s21 += h1 * h1;
      }
    }
  }

  if (!WRITE) {
    atomicAdd(&sred[c0], s10);
    atomicAdd(&sred[256 + c0], s20);
    atomicAdd(&sred[c1], s11);
    atomicAdd(&sred[256 + c1], s21);
    __syncthreads();
    atomicAdd(&statsum[tid], sred[tid]);
    atomicAdd(&statsum[256 + tid], sred[256 + tid]);
  }
}

// ---------------------------------------------------------------------------
// Kernel 4 (tiny): BN stats -> per-channel affine (scale, shift).
// ---------------------------------------------------------------------------
__global__ __launch_bounds__(256)
void finalize_kernel(const float* __restrict__ statsum, const float* __restrict__ gamma,
                     const float* __restrict__ beta, float* __restrict__ scale,
                     float* __restrict__ shiftv)
{
  const int c = threadIdx.x;
  const float inv_n = 1.f / (float)NPTS;
  const float mu  = statsum[c] * inv_n;
  const float m2  = statsum[256 + c] * inv_n;
  const float var = m2 - mu * mu;
  const float s   = gamma[c] * rsqrtf(var + BN_EPS);
  scale[c]  = s;
  shiftv[c] = fmaf(-mu, s, beta[c]);
}

// ---------------------------------------------------------------------------
extern "C" void kernel_launch(void* const* d_in, const int* in_sizes, int n_in,
                              void* d_out, int out_size, void* d_ws, size_t ws_size,
                              hipStream_t stream)
{
  const float* x     = (const float*)d_in[0];
  const float* w2    = (const float*)d_in[1];
  const float* b2    = (const float*)d_in[2];
  const float* w1    = (const float*)d_in[3];
  const float* b1    = (const float*)d_in[4];
  const float* gamma = (const float*)d_in[5];
  const float* beta  = (const float*)d_in[6];
  const int*   o     = (const int*)d_in[7];
  float* out = (float*)d_out;

  char* ws = (char*)d_ws;
  float* segsum  = (float*)(ws + WS_SEGSUM);
  float* statsum = (float*)(ws + WS_STATSUM);
  float* gg      = (float*)(ws + WS_GG);
  float* scale   = (float*)(ws + WS_SCALE);
  float* shiftv  = (float*)(ws + WS_SHIFT);
  bf16*  w1abf   = (bf16*)(ws + WS_W1ABF);
  bf16*  xbf     = (bf16*)(ws + WS_XBF);

  const bool cache = ws_size >= (size_t)WS_NEED_CACHE;

  // zero segment sums + BN stat accumulators (graph-capture safe)
  hipMemsetAsync(ws, 0, WS_GG, stream);

  const dim3 blk(256);
  const dim3 gSeg(NPTS / 128);
  const dim3 gGemm(NPTS / 64);

  if (cache) {
    segsum_kernel<true><<<gSeg, blk, 0, stream>>>(x, o, segsum, xbf);
    prep_kernel<true><<<1, blk, 0, stream>>>(segsum, w2, b2, w1, b1, o, gg, w1abf);
    gemm_pass<true, false><<<gGemm, blk, SMEM_BYTES, stream>>>(
        x, xbf, w1, w1abf, gg, o, scale, shiftv, statsum, out);
    finalize_kernel<<<1, blk, 0, stream>>>(statsum, gamma, beta, scale, shiftv);
    gemm_pass<true, true><<<gGemm, blk, SMEM_BYTES, stream>>>(
        x, xbf, w1, w1abf, gg, o, scale, shiftv, statsum, out);
  } else {
    segsum_kernel<false><<<gSeg, blk, 0, stream>>>(x, o, segsum, xbf);
    prep_kernel<false><<<1, blk, 0, stream>>>(segsum, w2, b2, w1, b1, o, gg, w1abf);
    gemm_pass<false, false><<<gGemm, blk, SMEM_BYTES, stream>>>(
        x, xbf, w1, w1abf, gg, o, scale, shiftv, statsum, out);
    finalize_kernel<<<1, blk, 0, stream>>>(statsum, gamma, beta, scale, shiftv);
    gemm_pass<false, true><<<gGemm, blk, SMEM_BYTES, stream>>>(
        x, xbf, w1, w1abf, gg, o, scale, shiftv, statsum, out);
  }
}
